// SpatialQFormerLayer_73263552135367
// MI455X (gfx1250) — matrix operand order repaired and checked by reading, not compiled, not run on hardware
//
#include <hip/hip_runtime.h>
#include <hip/hip_bf16.h>
#include <math.h>

// ---------------------------------------------------------------------------
// Types for CDNA5 WMMA (wave32, v_wmma_f32_16x16x32_bf16)
// ---------------------------------------------------------------------------
typedef __attribute__((ext_vector_type(16))) __bf16 v16bf;
typedef __attribute__((ext_vector_type(8)))  __bf16 v8bf;
typedef __attribute__((ext_vector_type(8)))  float  v8f;
typedef __attribute__((ext_vector_type(4)))  unsigned int v4u;
typedef __attribute__((ext_vector_type(8)))  unsigned int v8u;

#define DMODEL 1152
#define NROWS  46656          // B * N2 = 64 * 729
#define NGRP   576            // 64 * 9 window groups
#define HD     144            // head dim
#define TILE_M 64
#define TILE_N 128
#define TILE_K 32
#define LPAD   8              // LDS row padding (bf16 elems) -> 80B row stride

// ---------------------------------------------------------------------------
// fp32 -> bf16 elementwise (weights)
// ---------------------------------------------------------------------------
__global__ void sqf_f2bf_kernel(const float* __restrict__ src,
                                __bf16* __restrict__ dst, size_t n) {
  size_t i = (size_t)blockIdx.x * blockDim.x + threadIdx.x;
  size_t stride = (size_t)gridDim.x * blockDim.x;
  for (; i < n; i += stride) dst[i] = (__bf16)src[i];
}

// ---------------------------------------------------------------------------
// Tile query_token (9,1,D) -> (576, D) bf16
// ---------------------------------------------------------------------------
__global__ void sqf_tileq_kernel(const float* __restrict__ q,
                                 __bf16* __restrict__ dst) {
  int g = blockIdx.x;                       // 0..575
  const float* src = q + (size_t)(g % 9) * DMODEL;
  for (int j = threadIdx.x; j < DMODEL; j += 256)
    dst[(size_t)g * DMODEL + j] = (__bf16)src[j];
}

// ---------------------------------------------------------------------------
// Fused LayerNorm (optional) + torch-faithful window scatter, fp32 -> bf16.
//   B' = b*9 + r/3 ; F = (col/3)*9*D + 9*c + (r%3)*3 + (col%3)
// ---------------------------------------------------------------------------
__global__ __launch_bounds__(256)
void sqf_ln_window_kernel(const float* __restrict__ x,
                          const float* __restrict__ gam,
                          const float* __restrict__ bet,
                          __bf16* __restrict__ dst, int do_ln) {
  const int row = blockIdx.x;               // 0..46655
  const int tid = threadIdx.x;
  const float* xr = x + (size_t)row * DMODEL;

  float vals[5];
  float s = 0.f, s2 = 0.f;
#pragma unroll
  for (int it = 0; it < 5; ++it) {
    int j = tid + it * 256;
    float v = (j < DMODEL) ? xr[j] : 0.f;
    vals[it] = v;
    s += v; s2 += v * v;
  }
#pragma unroll
  for (int off = 16; off > 0; off >>= 1) {
    s  += __shfl_xor(s,  off, 32);
    s2 += __shfl_xor(s2, off, 32);
  }
  __shared__ float ws[8], ws2[8];
  if ((tid & 31) == 0) { ws[tid >> 5] = s; ws2[tid >> 5] = s2; }
  __syncthreads();
  float ts = 0.f, ts2 = 0.f;
#pragma unroll
  for (int i = 0; i < 8; ++i) { ts += ws[i]; ts2 += ws2[i]; }
  const float mean = ts * (1.f / DMODEL);
  const float var  = ts2 * (1.f / DMODEL) - mean * mean;
  const float rstd = rsqrtf(var + 1e-5f);

  const int b_i = row / 729;
  const int n   = row % 729;
  const int r   = n / 27, c = n % 27;
  const size_t base = (size_t)(b_i * 9 + r / 3) * 81 * DMODEL
                    + (size_t)(c / 3) * 9 * DMODEL
                    + (size_t)((r % 3) * 3 + (c % 3));
#pragma unroll
  for (int it = 0; it < 5; ++it) {
    int j = tid + it * 256;
    if (j < DMODEL) {
      float v = vals[it];
      if (do_ln) v = (v - mean) * rstd * gam[j] + bet[j];
      dst[base + (size_t)9 * j] = (__bf16)v;
    }
  }
}

// ---------------------------------------------------------------------------
// LayerNorm over 576 rows of D, fp32 -> bf16 row-major (norm2)
// ---------------------------------------------------------------------------
__global__ __launch_bounds__(256)
void sqf_ln_rows_kernel(const float* __restrict__ x,
                        const float* __restrict__ gam,
                        const float* __restrict__ bet,
                        __bf16* __restrict__ dst) {
  const int row = blockIdx.x;
  const int tid = threadIdx.x;
  const float* xr = x + (size_t)row * DMODEL;
  float vals[5];
  float s = 0.f, s2 = 0.f;
#pragma unroll
  for (int it = 0; it < 5; ++it) {
    int j = tid + it * 256;
    float v = (j < DMODEL) ? xr[j] : 0.f;
    vals[it] = v;
    s += v; s2 += v * v;
  }
#pragma unroll
  for (int off = 16; off > 0; off >>= 1) {
    s  += __shfl_xor(s,  off, 32);
    s2 += __shfl_xor(s2, off, 32);
  }
  __shared__ float ws[8], ws2[8];
  if ((tid & 31) == 0) { ws[tid >> 5] = s; ws2[tid >> 5] = s2; }
  __syncthreads();
  float ts = 0.f, ts2 = 0.f;
#pragma unroll
  for (int i = 0; i < 8; ++i) { ts += ws[i]; ts2 += ws2[i]; }
  const float mean = ts * (1.f / DMODEL);
  const float rstd = rsqrtf(ts2 * (1.f / DMODEL) - mean * mean + 1e-5f);
#pragma unroll
  for (int it = 0; it < 5; ++it) {
    int j = tid + it * 256;
    if (j < DMODEL)
      dst[(size_t)row * DMODEL + j] =
          (__bf16)((vals[it] - mean) * rstd * gam[j] + bet[j]);
  }
}

// ---------------------------------------------------------------------------
// bf16 WMMA GEMM with Tensor-Data-Mover tile staging + double-buffered LDS.
//   C(MxN) = A(MxK) * W(NxK)^T + bias[N]
//   mode 0: bias   mode 1: bias + exact GELU   mode 2: bias + residual
// Block: 128 threads = 4 waves (2x2), block tile 64x128, wave tile 32x64
// (2x4 WMMA). Requires M%64==0, N%128==0, K%32==0, tiles fully interior.
// TDM descriptor: 2-byte elems, 32-elem rows (64B), pad 4 DWORDs per 16-DWORD
// row -> 80B LDS row stride (== TILE_K + LPAD bf16).
// ---------------------------------------------------------------------------

// Launder a uniform value through an SGPR so downstream arithmetic never sees
// a ptrtoint(addrspacecast) ConstantExpr (works around an
// AMDGPUCodeGenPrepare constant-folding crash in clang-23).
__device__ __forceinline__ unsigned sqf_opaque_u32(unsigned x) {
  asm("" : "+s"(x));
  return x;
}

// D# group 1 (loop invariant): flags/dims/strides.
__device__ __forceinline__ v8u sqf_tdm_desc1(int rows, int K) {
  union { v8u v; unsigned long long q[4]; } g1;
  const unsigned long long td = 0x40000000ull;            // huge tensor dims
  // flags: data_size=2B, pad_enable, pad_interval=16DW(3), pad_amount=4DW(3)
  const unsigned long long flags =
      (1ull << 16) | (1ull << 20) | (3ull << 22) | (3ull << 25);
  g1.q[0] = flags | ((td & 0xFFFFull) << 48);             // tdim0[15:0]
  g1.q[1] = (td >> 16) | (td << 16) | (32ull << 48);      // tdim0hi|tdim1|tile0
  g1.q[2] = (unsigned long long)(unsigned)rows            // tile1 | tile2=0
          | ((unsigned long long)(unsigned)K << 32);      // stride0 = K elems
  g1.q[3] = 0ull;                                         // stride0hi|stride1
  return g1.v;
}

__device__ __forceinline__ void sqf_tdm_issue(unsigned lds_addr,
                                              unsigned long long gaddr,
                                              v8u g1) {
  union { v4u v; unsigned long long q[2]; } g0;
  g0.q[0] = 1ull | ((unsigned long long)lds_addr << 32);  // count=1 | lds_addr
  g0.q[1] = gaddr | (2ull << 62);                         // global_addr|type=2
  asm volatile("tensor_load_to_lds %0, %1" :: "s"(g0.v), "s"(g1) : "memory");
}

__global__ __launch_bounds__(128)
void sqf_gemm_bf16_wmma(const __bf16* __restrict__ A,
                        const __bf16* __restrict__ W,
                        const float* __restrict__ bias,
                        float* __restrict__ outF,
                        __bf16* __restrict__ outB,
                        const float* __restrict__ res,
                        int M, int N, int K, int mode) {
  __shared__ __align__(128) __bf16 lA[2][TILE_M][TILE_K + LPAD];
  __shared__ __align__(128) __bf16 lB[2][TILE_N][TILE_K + LPAD];

  const int tid  = threadIdx.x;
  const int lane = tid & 31;
  const int wave = tid >> 5;    // 0..3
  const int wm   = wave & 1;    // M slot of 32
  const int wn   = wave >> 1;   // N slot of 64
  const int bm   = blockIdx.y * TILE_M;
  const int bn   = blockIdx.x * TILE_N;

  const unsigned long long gA =
      (unsigned long long)(size_t)(A + (size_t)bm * K);
  const unsigned long long gB =
      (unsigned long long)(size_t)(W + (size_t)bn * K);
  const unsigned ldsA0 = sqf_opaque_u32((unsigned)(size_t)&lA[0][0][0]);
  const unsigned ldsA1 = sqf_opaque_u32((unsigned)(size_t)&lA[1][0][0]);
  const unsigned ldsB0 = sqf_opaque_u32((unsigned)(size_t)&lB[0][0][0]);
  const unsigned ldsB1 = sqf_opaque_u32((unsigned)(size_t)&lB[1][0][0]);
  const v8u g1A = sqf_tdm_desc1(TILE_M, K);
  const v8u g1B = sqf_tdm_desc1(TILE_N, K);

  v8f acc[2][4] = {};

  const int nk = K >> 5;                        // K / TILE_K
  if (wave == 0) {                              // prologue: stage buffer 0
    sqf_tdm_issue(ldsA0, gA, g1A);
    sqf_tdm_issue(ldsB0, gB, g1B);
  }

  for (int kt = 0; kt < nk; ++kt) {
    const int cur = kt & 1;
    __syncthreads();            // previous compute done -> next buf reusable
    if (wave == 0) {
      if (kt + 1 < nk) {
        const unsigned long long koff =
            (unsigned long long)(kt + 1) * (TILE_K * 2);
        sqf_tdm_issue(cur ? ldsA0 : ldsA1, gA + koff, g1A);
        sqf_tdm_issue(cur ? ldsB0 : ldsB1, gB + koff, g1B);
        __builtin_amdgcn_s_wait_tensorcnt(2);   // current buffer complete
      } else {
        __builtin_amdgcn_s_wait_tensorcnt(0);
      }
    }
    __syncthreads();            // current buffer visible to all waves

    // Fragment loads per documented 16-bit WMMA lane layouts
    const int mrow = lane & 15;
    const int kb   = (lane >> 4) * 8;    // A: K {kb..kb+7, 16+kb..16+kb+7}
    const int ks   = (lane >> 4) * 16;   // unused; B also takes K {kb..kb+7, 16+kb..16+kb+7}
    v16bf afrag[2], bfrag[4];
#pragma unroll
    for (int tm = 0; tm < 2; ++tm) {
      int r = wm * 32 + tm * 16 + mrow;
      v8bf lo = *(const v8bf*)&lA[cur][r][kb];
      v8bf hi = *(const v8bf*)&lA[cur][r][16 + kb];
      afrag[tm] = __builtin_shufflevector(lo, hi, 0, 1, 2, 3, 4, 5, 6, 7,
                                          8, 9, 10, 11, 12, 13, 14, 15);
    }
#pragma unroll
    for (int tn = 0; tn < 4; ++tn) {
      int cn = wn * 64 + tn * 16 + mrow;
      v8bf lo = *(const v8bf*)&lB[cur][cn][kb];
      v8bf hi = *(const v8bf*)&lB[cur][cn][16 + kb];
      bfrag[tn] = __builtin_shufflevector(lo, hi, 0, 1, 2, 3, 4, 5, 6, 7,
                                          8, 9, 10, 11, 12, 13, 14, 15);
    }
#pragma unroll
    for (int tm = 0; tm < 2; ++tm)
#pragma unroll
      for (int tn = 0; tn < 4; ++tn)
        acc[tm][tn] = __builtin_amdgcn_wmma_f32_16x16x32_bf16(
            false, afrag[tm], false, bfrag[tn], (short)0, acc[tm][tn],
            false, false);
  }

  // Epilogue: element i, lanes 0-15 -> M=i, N=lane; lanes 16-31 -> M=8+i
  const int ln = lane & 15;
  const int lm = (lane >> 4) * 8;
#pragma unroll
  for (int tm = 0; tm < 2; ++tm)
#pragma unroll
    for (int tn = 0; tn < 4; ++tn) {
      int col = bn + wn * 64 + tn * 16 + ln;
      float bv = bias ? bias[col] : 0.f;
#pragma unroll
      for (int i = 0; i < 8; ++i) {
        int rowo = bm + wm * 32 + tm * 16 + lm + i;
        float v = acc[tm][tn][i] + bv;
        if (mode == 1) v = 0.5f * v * (1.0f + erff(v * 0.70710678118654752f));
        size_t off = (size_t)rowo * N + col;
        if (mode == 2) v += res[off];
        if (outF) outF[off] = v;
        if (outB) outB[off] = (__bf16)v;
      }
    }
}

// ---------------------------------------------------------------------------
// Per-group single-query attention: Q (576,1152) f32, KV (576*81, 2304) bf16
// (cols 0..1151 = K proj, 1152..2303 = V proj). One block per group, one wave
// per head. ctx out (576,1152) bf16 = concat-head context (pre out-proj).
// ---------------------------------------------------------------------------
__global__ __launch_bounds__(256)
void sqf_attn_kernel(const float* __restrict__ Q,
                     const __bf16* __restrict__ KV,
                     __bf16* __restrict__ ctx) {
  const int g    = blockIdx.x;   // 0..575
  const int tid  = threadIdx.x;
  const int lane = tid & 31;
  const int h    = tid >> 5;     // head 0..7
  __shared__ float prob[8][88];

  const float*  q   = Q + (size_t)g * DMODEL + h * HD;
  const __bf16* kvg = KV + (size_t)g * 81 * 2304;

  float sc[3];
#pragma unroll
  for (int it = 0; it < 3; ++it) {
    int k = lane + it * 32;
    float dot = -1e30f;
    if (k < 81) {
      const __bf16* kr = kvg + (size_t)k * 2304 + h * HD;
      float acc = 0.f;
      for (int d = 0; d < HD; ++d) acc += q[d] * (float)kr[d];
      dot = acc * (1.0f / 12.0f);          // 1/sqrt(144)
    }
    sc[it] = dot;
  }
  float m = fmaxf(sc[0], fmaxf(sc[1], sc[2]));
#pragma unroll
  for (int off = 16; off > 0; off >>= 1) m = fmaxf(m, __shfl_xor(m, off, 32));
  float sum = 0.f;
#pragma unroll
  for (int it = 0; it < 3; ++it) {
    sc[it] = (lane + it * 32 < 81) ? __expf(sc[it] - m) : 0.f;
    sum += sc[it];
  }
#pragma unroll
  for (int off = 16; off > 0; off >>= 1) sum += __shfl_xor(sum, off, 32);
  const float rs = 1.f / sum;
#pragma unroll
  for (int it = 0; it < 3; ++it) {
    int k = lane + it * 32;
    if (k < 81) prob[h][k] = sc[it] * rs;
  }
  __syncthreads();

#pragma unroll
  for (int it = 0; it < 5; ++it) {
    int d = lane + it * 32;
    if (d < HD) {
      const __bf16* vr = kvg + 1152 + h * HD + d;
      float acc = 0.f;
      for (int k = 0; k < 81; ++k)
        acc += prob[h][k] * (float)vr[(size_t)k * 2304];
      ctx[(size_t)g * DMODEL + h * HD + d] = (__bf16)acc;
    }
  }
}

// ---------------------------------------------------------------------------
// Host orchestration
// ---------------------------------------------------------------------------
static inline void sqf_gemm(const __bf16* A, const __bf16* W, const float* b,
                            float* oF, __bf16* oB, const float* res,
                            int M, int N, int K, int mode, hipStream_t s) {
  dim3 grid(N / TILE_N, M / TILE_M);
  sqf_gemm_bf16_wmma<<<grid, 128, 0, s>>>(A, W, b, oF, oB, res, M, N, K, mode);
}

extern "C" void kernel_launch(void* const* d_in, const int* in_sizes, int n_in,
                              void* d_out, int out_size, void* d_ws,
                              size_t ws_size, hipStream_t stream) {
  (void)in_sizes; (void)n_in; (void)out_size; (void)ws_size;
  const float* x1       = (const float*)d_in[0];
  const float* x2       = (const float*)d_in[1];
  const float* emb      = (const float*)d_in[2];
  const float* qtok     = (const float*)d_in[3];
  const float* n1g      = (const float*)d_in[4];
  const float* n1b      = (const float*)d_in[5];
  const float* a1_in_w  = (const float*)d_in[6];
  const float* a1_in_b  = (const float*)d_in[7];
  const float* a1_out_w = (const float*)d_in[8];
  const float* a1_out_b = (const float*)d_in[9];
  const float* a2_in_w  = (const float*)d_in[10];
  const float* a2_in_b  = (const float*)d_in[11];
  const float* a2_out_w = (const float*)d_in[12];
  const float* a2_out_b = (const float*)d_in[13];
  const float* n2g      = (const float*)d_in[14];
  const float* n2b      = (const float*)d_in[15];
  const float* mlp_w1   = (const float*)d_in[16];
  const float* mlp_b1   = (const float*)d_in[17];
  const float* mlp_w2   = (const float*)d_in[18];
  const float* mlp_b2   = (const float*)d_in[19];

  float* out = (float*)d_out;
  const size_t OUT_BR = (size_t)NGRP * DMODEL;          // 663552

  // workspace carving
  size_t off = 0;
  char* ws = (char*)d_ws;
  auto take = [&](size_t elems, size_t esz) -> void* {
    void* p = ws + off;
    off += (elems * esz + 255) & ~(size_t)255;
    return p;
  };
  const size_t DD = (size_t)DMODEL * DMODEL;
  __bf16* wA1in  = (__bf16*)take(3 * DD, 2);
  __bf16* wA2in  = (__bf16*)take(3 * DD, 2);
  __bf16* wA1out = (__bf16*)take(DD, 2);
  __bf16* wA2out = (__bf16*)take(DD, 2);
  __bf16* wM1    = (__bf16*)take(2 * DD, 2);
  __bf16* wM2    = (__bf16*)take(2 * DD, 2);
  __bf16* qin    = (__bf16*)take((size_t)NGRP * DMODEL, 2);
  __bf16* actw   = (__bf16*)take((size_t)NROWS * DMODEL, 2);
  __bf16* kvb    = (__bf16*)take((size_t)NROWS * 2 * DMODEL, 2);
  __bf16* ctxb   = (__bf16*)take((size_t)NGRP * DMODEL, 2);
  __bf16* aembb  = (__bf16*)take((size_t)NGRP * DMODEL, 2);
  __bf16* nbf    = (__bf16*)take((size_t)NGRP * DMODEL, 2);
  __bf16* hbf    = (__bf16*)take((size_t)NGRP * 2 * DMODEL, 2);
  float*  Q1     = (float*)take((size_t)NGRP * DMODEL, 4);
  float*  Q2     = (float*)take((size_t)NGRP * DMODEL, 4);
  float*  ares   = (float*)take((size_t)NGRP * DMODEL, 4);

  // 1) weights -> bf16
  sqf_f2bf_kernel<<<2048, 256, 0, stream>>>(a1_in_w, wA1in, 3 * DD);
  sqf_f2bf_kernel<<<2048, 256, 0, stream>>>(a2_in_w, wA2in, 3 * DD);
  sqf_f2bf_kernel<<<2048, 256, 0, stream>>>(a1_out_w, wA1out, DD);
  sqf_f2bf_kernel<<<2048, 256, 0, stream>>>(a2_out_w, wA2out, DD);
  sqf_f2bf_kernel<<<2048, 256, 0, stream>>>(mlp_w1, wM1, 2 * DD);
  sqf_f2bf_kernel<<<2048, 256, 0, stream>>>(mlp_w2, wM2, 2 * DD);
  sqf_tileq_kernel<<<NGRP, 256, 0, stream>>>(qtok, qin);

  // 2) attn1: window(embedding), Q1, KV1, attention, out-proj -> attn_emb(bf16)
  sqf_ln_window_kernel<<<NROWS, 256, 0, stream>>>(emb, nullptr, nullptr, actw, 0);
  sqf_gemm(qin, wA1in, a1_in_b, Q1, nullptr, nullptr,
           NGRP, DMODEL, DMODEL, 0, stream);
  sqf_gemm(actw, wA1in + DD, a1_in_b + DMODEL, nullptr, kvb, nullptr,
           NROWS, 2 * DMODEL, DMODEL, 0, stream);
  sqf_attn_kernel<<<NGRP, 256, 0, stream>>>(Q1, kvb, ctxb);
  sqf_gemm(ctxb, wA1out, a1_out_b, nullptr, aembb, nullptr,
           NGRP, DMODEL, DMODEL, 0, stream);

  // 3) Q2 = attn_emb @ attn2_in_w[:D]^T + b (shared by both branches)
  sqf_gemm(aembb, wA2in, a2_in_b, Q2, nullptr, nullptr,
           NGRP, DMODEL, DMODEL, 0, stream);

  // 4) branches
  const float* xs[2] = {x1, x2};
  for (int br = 0; br < 2; ++br) {
    sqf_ln_window_kernel<<<NROWS, 256, 0, stream>>>(xs[br], n1g, n1b, actw, 1);
    sqf_gemm(actw, wA2in + DD, a2_in_b + DMODEL, nullptr, kvb, nullptr,
             NROWS, 2 * DMODEL, DMODEL, 0, stream);
    sqf_attn_kernel<<<NGRP, 256, 0, stream>>>(Q2, kvb, ctxb);
    sqf_gemm(ctxb, wA2out, a2_out_b, ares, nullptr, nullptr,
             NGRP, DMODEL, DMODEL, 0, stream);
    sqf_ln_rows_kernel<<<NGRP, 256, 0, stream>>>(ares, n2g, n2b, nbf);
    sqf_gemm(nbf, wM1, mlp_b1, nullptr, hbf, nullptr,
             NGRP, 2 * DMODEL, DMODEL, 1, stream);                 // GELU
    sqf_gemm(hbf, wM2, mlp_b2, out + (size_t)br * OUT_BR, nullptr, ares,
             NGRP, DMODEL, 2 * DMODEL, 2, stream);                 // + residual
  }

  // 5) embedding passthrough
  (void)hipMemcpyAsync(out + 2 * OUT_BR, emb,
                       (size_t)NROWS * DMODEL * sizeof(float),
                       hipMemcpyDeviceToDevice, stream);
}
